// VMambaBasicBackbone_42107859370359
// MI455X (gfx1250) — matrix operand order repaired and checked by reading, not compiled, not run on hardware
//
#include <hip/hip_runtime.h>
#include <math.h>

// MI455X / gfx1250, wave32. bf16 WMMA (v_wmma_f32_16x16x32_bf16) for all
// channel-mixing GEMMs, fp32 VALU for LN/scan/dwconv/gating epilogues.

typedef __attribute__((ext_vector_type(16))) __bf16 v16bf;
typedef __attribute__((ext_vector_type(8)))  __bf16 v8bf;
typedef __attribute__((ext_vector_type(8)))  float  v8f;

__device__ __forceinline__ float gelu_exact(float v) {
    return 0.5f * v * (1.0f + erff(v * 0.70710678118654752f));
}
__device__ __forceinline__ float sigm(float v) { return 1.0f / (1.0f + expf(-v)); }

// ---------------------------------------------------------------------------
// GEMM: Y[n,o,p] = sum_k W[o,k]*X[n,k,p]  (+bias[o]) (gelu?) (+res[n,o,p])
// Requires K%32==0, P%64==0, O%16==0 (true for every layer in this net).
// Block = 128 threads (4 waves); block tile = 64(M) x 64(N); wave = 16x64.
// B tile staged in LDS TRANSPOSED as bf16 (BsT[col][k], row padded to 40
// shorts) so each lane's fragment = 16 consecutive shorts -> 2x ds_load_b128.
// Global loads for tile k+32 are issued before the WMMAs on tile k (software
// pipeline); prefetch pulls tile k+64. All four B fragments are preloaded so
// the four v_wmma issue back-to-back (XDL-friendly, §7.5).
// ---------------------------------------------------------------------------
__global__ __launch_bounds__(128) void k_gemm_wmma(
    const float* __restrict__ Wm, const float* __restrict__ Xm,
    const float* __restrict__ bias, const float* __restrict__ res,
    float* __restrict__ Y, int O, int K, int P, int doGelu)
{
    __shared__ __bf16 BsT[64][40];          // 5 KB; 80B rows: b128-aligned frags
    const long nK = (long)blockIdx.z * K * P;
    const long nO = (long)blockIdx.z * O * P;
    const float* Xn = Xm + nK;
    const float* Rn = res ? res + nO : nullptr;
    float* Yn = Y + nO;

    const int tid  = threadIdx.x;
    const int wv   = tid >> 5;
    const int lane = tid & 31;
    const int l15  = lane & 15;
    const int hi   = lane >> 4;

    const int tileP = blockIdx.x * 64;
    const int waveM = blockIdx.y * 64 + wv * 16;
    const bool active = (waveM < O);

    const int lc  = tid & 63;               // column this thread stages
    const int lr0 = tid >> 6;               // row-pair phase (0 or 1)

    v8f acc[4] = {};
    float ra[8], rb[8];                     // staged K-row pair per column

    // prologue: load + store tile k=0
    #pragma unroll
    for (int i = 0; i < 8; ++i) {
        int rp = lr0 + 2 * i;
        ra[i] = Xn[(long)(2 * rp)     * P + tileP + lc];
        rb[i] = Xn[(long)(2 * rp + 1) * P + tileP + lc];
    }
    #pragma unroll
    for (int i = 0; i < 8; ++i) {
        int rp = lr0 + 2 * i;
        BsT[lc][2 * rp]     = (__bf16)ra[i];
        BsT[lc][2 * rp + 1] = (__bf16)rb[i];
    }
    __syncthreads();

    for (int k = 0; k < K; k += 32) {
        if (k + 32 < K) {                   // issue next tile's global loads now
            #pragma unroll
            for (int i = 0; i < 8; ++i) {
                int rp = lr0 + 2 * i;
                ra[i] = Xn[(long)(k + 32 + 2 * rp)     * P + tileP + lc];
                rb[i] = Xn[(long)(k + 32 + 2 * rp + 1) * P + tileP + lc];
            }
            if (k + 64 < K)                 // speculative prefetch of tile k+64
                __builtin_prefetch(&Xn[(long)(k + 64 + lr0) * P + tileP + lc], 0, 3);
        }

        if (active) {
            // A fragment, this code's contiguous k order (k = 16*hi + j): lanes0-15: K 0..15,
            // lanes16-31: K 16..31 (row M = lane&15).
            v16bf a;
            const float* wr = Wm + (long)(waveM + l15) * K + k + hi * 16;
            #pragma unroll
            for (int j = 0; j < 8; ++j) a[j]     = (__bf16)wr[j];
            #pragma unroll
            for (int j = 0; j < 8; ++j) a[8 + j] = (__bf16)wr[8 + j];

            // Preload all four B fragments, then issue WMMAs back-to-back.
            v16bf bfr[4];
            #pragma unroll
            for (int t = 0; t < 4; ++t) {
                // B fragment, same contiguous k order as A: lanes0-15 K 0..15, lanes16-31
                // K 16..31, N = lane&15. Consecutive in BsT -> 2x ds_load_b128.
                const v8bf* bp = (const v8bf*)&BsT[t * 16 + l15][hi * 16];
                v8bf b0 = bp[0], b1 = bp[1];
                bfr[t] = __builtin_shufflevector(b0, b1,
                    0, 1, 2, 3, 4, 5, 6, 7, 8, 9, 10, 11, 12, 13, 14, 15);
            }
            #pragma unroll
            for (int t = 0; t < 4; ++t)
                acc[t] = __builtin_amdgcn_wmma_f32_16x16x32_bf16(
                    false, a, false, bfr[t], (short)0, acc[t], false, false);
        }
        __syncthreads();

        if (k + 32 < K) {                   // publish next tile
            #pragma unroll
            for (int i = 0; i < 8; ++i) {
                int rp = lr0 + 2 * i;
                BsT[lc][2 * rp]     = (__bf16)ra[i];
                BsT[lc][2 * rp + 1] = (__bf16)rb[i];
            }
        }
        __syncthreads();
    }

    if (!active) return;
    // C/D 16x16 f32 layout: VGPR r -> M = r + (lane>=16 ? 8 : 0), N = lane&15.
    #pragma unroll
    for (int t = 0; t < 4; ++t) {
        #pragma unroll
        for (int r = 0; r < 8; ++r) {
            int row = waveM + r + hi * 8;
            int col = tileP + t * 16 + l15;
            float v = acc[t][r];
            if (bias) v += bias[row];
            if (doGelu) v = gelu_exact(v);
            long o = (long)row * P + col;
            if (Rn) v += Rn[o];
            Yn[o] = v;
        }
    }
}

// ---------------------------------------------------------------------------
// Space-to-depth: (N,C,H,W) -> (N, C*s*s, H/s, W/s); channel (c*s+dy)*s+dx.
// Turns 4x4/s4 patch-embed and 2x2/s2 merges into plain GEMMs (OIHW weights
// flatten to exactly this K order).
// ---------------------------------------------------------------------------
__global__ void k_s2d(const float* __restrict__ in, float* __restrict__ out,
                      int C, int H, int W, int s, long total)
{
    long id = (long)blockIdx.x * blockDim.x + threadIdx.x;
    if (id >= total) return;
    int Wo = W / s, Ho = H / s, Co = C * s * s;
    long t = id;
    int j = (int)(t % Wo); t /= Wo;
    int i = (int)(t % Ho); t /= Ho;
    int co = (int)(t % Co); t /= Co;
    int n = (int)t;
    int dx = co % s; int r = co / s;
    int dy = r % s;  int c = r / s;
    out[id] = in[(((long)n * C + c) * H + (long)i * s + dy) * W + (long)j * s + dx];
}

// ---------------------------------------------------------------------------
// Channel LayerNorm per (n,h,w).
// ---------------------------------------------------------------------------
__global__ void k_ln2d(const float* __restrict__ in, const float* __restrict__ w,
                       const float* __restrict__ b, float* __restrict__ out,
                       int C, int HW, long total)
{
    long id = (long)blockIdx.x * blockDim.x + threadIdx.x;
    if (id >= total) return;
    int sp = (int)(id % HW); int n = (int)(id / HW);
    const float* base = in + (long)n * C * HW + sp;
    float s = 0.f, s2 = 0.f;
    for (int c = 0; c < C; ++c) { float v = base[(long)c * HW]; s += v; s2 += v * v; }
    float m = s / C;
    float inv = rsqrtf(fmaxf(s2 / C - m * m, 0.f) + 1e-6f);
    float* ob = out + (long)n * C * HW + sp;
    for (int c = 0; c < C; ++c)
        ob[(long)c * HW] = (base[(long)c * HW] - m) * inv * w[c] + b[c];
}

// ---------------------------------------------------------------------------
// Depthwise 3x3 SAME, no bias. Input may live inside a wider tensor (inC).
// ---------------------------------------------------------------------------
__global__ void k_dw3x3(const float* __restrict__ in, const float* __restrict__ w,
                        float* __restrict__ out, int C, int inC, int H, int W, long total)
{
    long id = (long)blockIdx.x * blockDim.x + threadIdx.x;
    if (id >= total) return;
    int x = (int)(id % W); long t = id / W;
    int y = (int)(t % H); t /= H;
    int c = (int)(t % C); int n = (int)(t / C);
    const float* ip = in + ((long)n * inC + c) * H * W;
    const float* wc = w + c * 9;
    float acc = 0.f;
    #pragma unroll
    for (int ky = 0; ky < 3; ++ky) {
        int yy = y + ky - 1; if (yy < 0 || yy >= H) continue;
        #pragma unroll
        for (int kx = 0; kx < 3; ++kx) {
            int xx = x + kx - 1; if (xx < 0 || xx >= W) continue;
            acc += wc[ky * 3 + kx] * ip[(long)yy * W + xx];
        }
    }
    out[id] = acc;
}

// ---------------------------------------------------------------------------
// Per-block scan parameter prep: prm[0..4C)=clip(sigmoid(decay)),
// prm[4C..8C)=1+tanh(iscale), prm[8C..12C)=softmax(mix over dirs).
// ---------------------------------------------------------------------------
__global__ void k_prep(const float* __restrict__ decay, const float* __restrict__ mix,
                       const float* __restrict__ iscale, float* __restrict__ prm, int C)
{
    int c = blockIdx.x * blockDim.x + threadIdx.x;
    if (c >= C) return;
    float mx = -1e30f;
    #pragma unroll
    for (int d = 0; d < 4; ++d) {
        float dv = 1.f / (1.f + expf(-decay[d * C + c]));
        prm[d * C + c] = fminf(fmaxf(dv, 0.05f), 0.995f);
        prm[(4 + d) * C + c] = 1.f + tanhf(iscale[d * C + c]);
        mx = fmaxf(mx, mix[d * C + c]);
    }
    float e[4]; float sum = 0.f;
    #pragma unroll
    for (int d = 0; d < 4; ++d) { e[d] = expf(mix[d * C + c] - mx); sum += e[d]; }
    #pragma unroll
    for (int d = 0; d < 4; ++d) prm[(8 + d) * C + c] = e[d] / sum;
}

// Horizontal EMA scans (dirs 0=lr, 1=rl): out = m0*lr + m1*rl.
__global__ void k_scan_h(const float* __restrict__ x, const float* __restrict__ prm,
                         float* __restrict__ out, int C, int H, int W, long total)
{
    long id = (long)blockIdx.x * blockDim.x + threadIdx.x;
    if (id >= total) return;
    int h = (int)(id % H); long t = id / H;
    int c = (int)(t % C); int n = (int)(t / C);
    long base = (((long)n * C + c) * H + h) * (long)W;
    const float* row = x + base;
    float* orow = out + base;
    float d0 = prm[c],         d1 = prm[C + c];
    float s0 = prm[4 * C + c], s1 = prm[5 * C + c];
    float m0 = prm[8 * C + c], m1 = prm[9 * C + c];
    float st = 0.f;
    for (int wj = 0; wj < W; ++wj) { st = d0 * st + (1.f - d0) * (row[wj] * s0); orow[wj] = m0 * st; }
    st = 0.f;
    for (int wj = W - 1; wj >= 0; --wj) { st = d1 * st + (1.f - d1) * (row[wj] * s1); orow[wj] += m1 * st; }
}

// Vertical EMA scans (dirs 2=tb, 3=bt), coalesced (consecutive threads = w).
__global__ void k_scan_v(const float* __restrict__ x, const float* __restrict__ prm,
                         float* __restrict__ out, int C, int H, int W, long total)
{
    long id = (long)blockIdx.x * blockDim.x + threadIdx.x;
    if (id >= total) return;
    int wc = (int)(id % W); long t = id / W;
    int c = (int)(t % C); int n = (int)(t / C);
    long base = (((long)n * C + c) * H) * (long)W + wc;
    const float* colp = x + base;
    float* ocol = out + base;
    float d2 = prm[2 * C + c],  d3 = prm[3 * C + c];
    float s2 = prm[6 * C + c],  s3 = prm[7 * C + c];
    float m2 = prm[10 * C + c], m3 = prm[11 * C + c];
    float st = 0.f;
    for (int h = 0; h < H; ++h)     { st = d2 * st + (1.f - d2) * (colp[(long)h * W] * s2); ocol[(long)h * W] += m2 * st; }
    st = 0.f;
    for (int h = H - 1; h >= 0; --h){ st = d3 * st + (1.f - d3) * (colp[(long)h * W] * s3); ocol[(long)h * W] += m3 * st; }
}

// Fused: local = dw3x3(proj, le_w); v = (scan + local) * sigmoid(gate);
// BN (inference); exact GELU.
__global__ void k_fuse(const float* __restrict__ scan, const float* __restrict__ proj,
                       const float* __restrict__ lew, const float* __restrict__ pj,
                       const float* __restrict__ bnw, const float* __restrict__ bnb,
                       const float* __restrict__ bnm, const float* __restrict__ bnv,
                       float* __restrict__ out, int C, int H, int W, long total)
{
    long id = (long)blockIdx.x * blockDim.x + threadIdx.x;
    if (id >= total) return;
    int x = (int)(id % W); long t = id / W;
    int y = (int)(t % H); t /= H;
    int c = (int)(t % C); int n = (int)(t / C);
    const float* ip = proj + ((long)n * C + c) * H * W;
    const float* wc = lew + c * 9;
    float local = 0.f;
    #pragma unroll
    for (int ky = 0; ky < 3; ++ky) {
        int yy = y + ky - 1; if (yy < 0 || yy >= H) continue;
        #pragma unroll
        for (int kx = 0; kx < 3; ++kx) {
            int xx = x + kx - 1; if (xx < 0 || xx >= W) continue;
            local += wc[ky * 3 + kx] * ip[(long)yy * W + xx];
        }
    }
    float g = sigm(pj[((long)n * 2 * C + C + c) * H * W + (long)y * W + x]);
    float v = (scan[id] + local) * g;
    v = (v - bnm[c]) * rsqrtf(bnv[c] + 1e-5f) * bnw[c] + bnb[c];
    out[id] = gelu_exact(v);
}

// ---------------------------------------------------------------------------
extern "C" void kernel_launch(void* const* d_in, const int* in_sizes, int n_in,
                              void* d_out, int out_size, void* d_ws, size_t ws_size,
                              hipStream_t stream)
{
    const int DIMS[4] = {96, 192, 384, 768};
    const int DEPTHS[4] = {2, 2, 6, 2};
    const int N = 2;

    const float* x_in = (const float*)d_in[0];
    auto PP = [&](int i) { return (const float*)d_in[i]; };
    // Assumed flatten order after x (sorted-key pytree, JAX convention):
    // [1..9]  merges: 3 x {n_b, n_w, w}
    // [10] pe_n_b  [11] pe_n_w  [12] pe_w
    // [13..] stages: 12 blocks x 21 tensors, keys alphabetical:
    enum {BN_B=0,BN_M,BN_V,BN_W,DECAY,DW_W,FC1_B,FC1_W,FC2_B,FC2_W,
          IN_B,IN_W,ISCALE,LE_W,MIX,N1_B,N1_W,N2_B,N2_W,OUT_B,OUT_W};

    float* ws   = (float*)d_ws;             // needs ~25.5M floats (~102 MB)
    float* bufX = ws;                       // current x / residual (<=3.15M)
    float* bufA = ws + 3200000;             // normed / s2d / mixed
    float* bufC = ws + 6400000;             // projected (dw) / merge out
    float* bufD = ws + 9600000;             // scanned
    float* bufB = ws + 12800000;            // pj (2d) / fc1 hidden (4d), <=12.6M
    float* prm  = ws + 25400000;            // 12*768 scan params

    auto cdiv = [](long a, long b) { return (int)((a + b - 1) / b); };

    // ---- patch embed: s2d(4) + GEMM(96x64) + LN ----
    {
        long tot = (long)N * 64 * 128 * 128;
        k_s2d<<<cdiv(tot, 256), 256, 0, stream>>>(x_in, bufA, 4, 512, 512, 4, tot);
        k_gemm_wmma<<<dim3(16384 / 64, cdiv(96, 64), N), 128, 0, stream>>>(
            PP(12), bufA, nullptr, nullptr, bufC, 96, 64, 16384, 0);
        long lt = (long)N * 16384;
        k_ln2d<<<cdiv(lt, 256), 256, 0, stream>>>(bufC, PP(11), PP(10), bufX, 96, 16384, lt);
    }

    int stageBase[4];
    { int a = 0; for (int s = 0; s < 4; ++s) { stageBase[s] = 13 + a * 21; a += DEPTHS[s]; } }

    long outOff = 0;
    int H = 128, W = 128;
    for (int s = 0; s < 4; ++s) {
        const int d = DIMS[s];
        const int HW = H * W;
        const long tNCHW = (long)N * d * HW;
        const long tPos  = (long)N * HW;
        for (int b = 0; b < DEPTHS[s]; ++b) {
            auto T = [&](int k) { return PP(stageBase[s] + b * 21 + k); };
            // SSM branch
            k_ln2d<<<cdiv(tPos, 256), 256, 0, stream>>>(bufX, T(N1_W), T(N1_B), bufA, d, HW, tPos);
            k_gemm_wmma<<<dim3(HW / 64, cdiv(2 * d, 64), N), 128, 0, stream>>>(
                T(IN_W), bufA, T(IN_B), nullptr, bufB, 2 * d, d, HW, 0);
            k_dw3x3<<<cdiv(tNCHW, 256), 256, 0, stream>>>(bufB, T(DW_W), bufC, d, 2 * d, H, W, tNCHW);
            k_prep<<<cdiv(d, 128), 128, 0, stream>>>(T(DECAY), T(MIX), T(ISCALE), prm, d);
            { long th = (long)N * d * H;
              k_scan_h<<<cdiv(th, 128), 128, 0, stream>>>(bufC, prm, bufD, d, H, W, th); }
            { long tv = (long)N * d * W;
              k_scan_v<<<cdiv(tv, 128), 128, 0, stream>>>(bufC, prm, bufD, d, H, W, tv); }
            k_fuse<<<cdiv(tNCHW, 256), 256, 0, stream>>>(
                bufD, bufC, T(LE_W), bufB, T(BN_W), T(BN_B), T(BN_M), T(BN_V), bufA, d, H, W, tNCHW);
            k_gemm_wmma<<<dim3(HW / 64, cdiv(d, 64), N), 128, 0, stream>>>(
                T(OUT_W), bufA, T(OUT_B), bufX, bufX, d, d, HW, 0);
            // MLP branch
            k_ln2d<<<cdiv(tPos, 256), 256, 0, stream>>>(bufX, T(N2_W), T(N2_B), bufA, d, HW, tPos);
            k_gemm_wmma<<<dim3(HW / 64, cdiv(4 * d, 64), N), 128, 0, stream>>>(
                T(FC1_W), bufA, T(FC1_B), nullptr, bufB, 4 * d, d, HW, 1);
            k_gemm_wmma<<<dim3(HW / 64, cdiv(d, 64), N), 128, 0, stream>>>(
                T(FC2_W), bufB, T(FC2_B), bufX, bufX, d, 4 * d, HW, 0);
        }
        hipMemcpyAsync((float*)d_out + outOff, bufX, tNCHW * sizeof(float),
                       hipMemcpyDeviceToDevice, stream);
        outOff += tNCHW;
        if (s < 3) {  // patch merge: s2d(2) + GEMM + LN (H,W even -> no pad)
            const int d2 = DIMS[s + 1];
            const int Ho = H / 2, Wo = W / 2, HWo = Ho * Wo;
            long tot = (long)N * 4 * d * HWo;
            k_s2d<<<cdiv(tot, 256), 256, 0, stream>>>(bufX, bufA, d, H, W, 2, tot);
            k_gemm_wmma<<<dim3(HWo / 64, cdiv(d2, 64), N), 128, 0, stream>>>(
                PP(1 + s * 3 + 2), bufA, nullptr, nullptr, bufC, d2, 4 * d, HWo, 0);
            long lt = (long)N * HWo;
            k_ln2d<<<cdiv(lt, 256), 256, 0, stream>>>(
                bufC, PP(1 + s * 3 + 1), PP(1 + s * 3 + 0), bufX, d2, HWo, lt);
            H = Ho; W = Wo;
        }
    }
    (void)in_sizes; (void)n_in; (void)out_size; (void)ws_size;
}
